// Decoder_49469433315669
// MI455X (gfx1250) — compile-verified, matrix operand order checked by reading, not run on hardware
//
#include <hip/hip_runtime.h>
#include <hip/hip_bf16.h>
#include <math.h>

namespace {

constexpr int S = 1024;
constexpr int E = 4096;
constexpr int H = 32;
constexpr int D = 128;
constexpr float GAMMA = 64.0f;
constexpr float SCALE = 0.08838834764831845f;  // 1/sqrt(128)

typedef __attribute__((ext_vector_type(16))) __bf16 v16bf;
typedef __attribute__((ext_vector_type(8)))  __bf16 v8bf;
typedef __attribute__((ext_vector_type(8)))  float  v8f;
typedef __attribute__((ext_vector_type(4)))  float  v4f;

#define DEVI static __device__ __forceinline__

// ---------------------------------------------------------------------------
// WMMA fragment loaders.
// CDNA5 16-bit A-frag (16x32): lane L holds row (L&15); K values per lane:
//   elements 0..7  -> k0 + (L>>4)*8 + 0..7        (contiguous)
//   elements 8..15 -> k0 + 16 + (L>>4)*8 + 0..7   (contiguous)
// B-frag (32x16) is lane-symmetric: lane holds column (L&15) of B, i.e. a
// row of the row-major B^T source, same K striping. So both frags are two
// 16-byte (bf16) / two 32-byte (f32, converted) contiguous chunks per lane.
// ---------------------------------------------------------------------------

DEVI v16bf frag_load_f32(const float* p) {
  v4f a0 = *(const v4f*)(p + 0);
  v4f a1 = *(const v4f*)(p + 4);
  v4f b0 = *(const v4f*)(p + 16);
  v4f b1 = *(const v4f*)(p + 20);
  v16bf f;
#pragma unroll
  for (int i = 0; i < 4; ++i) {
    f[i]      = (__bf16)a0[i];
    f[4 + i]  = (__bf16)a1[i];
    f[8 + i]  = (__bf16)b0[i];
    f[12 + i] = (__bf16)b1[i];
  }
  return f;
}

DEVI v16bf frag_load_bf16(const __bf16* p) {
  v8bf a = *(const v8bf*)(p + 0);
  v8bf b = *(const v8bf*)(p + 16);
  v16bf f;
#pragma unroll
  for (int i = 0; i < 8; ++i) { f[i] = a[i]; f[8 + i] = b[i]; }
  return f;
}

// C/D tile layout: VGPR r holds row (8*(L>>4) + r), column (L&15).
DEVI void store_tile(float* dst, size_t ld, int row0, int col0, int lane,
                     const v8f& acc) {
  const int n = lane & 15, g = lane >> 4;
#pragma unroll
  for (int r = 0; r < 8; ++r)
    dst[(size_t)(row0 + 8 * g + r) * ld + col0 + n] = acc[r];
}

// D = A(row-major, MxK) * B^T(row-major, NxK), one 16x(16*NT) strip per wave.
template <int NT>
DEVI void gemm_strip_f32(const float* A, int lda, int row0,
                         const float* Bt, int ldb, int col0,
                         int K, int lane, v8f* acc) {
  const int mr = lane & 15, g = lane >> 4;
  const float* arow = A + (size_t)(row0 + mr) * lda + g * 8;
  const float* brow[NT];
#pragma unroll
  for (int t = 0; t < NT; ++t)
    brow[t] = Bt + (size_t)(col0 + 16 * t + mr) * ldb + g * 8;
  for (int k0 = 0; k0 < K; k0 += 32) {
    v16bf a = frag_load_f32(arow + k0);
    if (k0 + 32 < K) __builtin_prefetch(arow + k0 + 32, 0, 0);
#pragma unroll
    for (int t = 0; t < NT; ++t) {
      v16bf b = frag_load_f32(brow[t] + k0);
      acc[t] = __builtin_amdgcn_wmma_f32_16x16x32_bf16(
          false, a, false, b, (short)0, acc[t], false, false);
    }
  }
}

template <int NT>
DEVI void gemm_strip_bf16(const __bf16* A, int lda, int row0,
                          const __bf16* Bt, int ldb, int col0,
                          int K, int lane, v8f* acc) {
  const int mr = lane & 15, g = lane >> 4;
  const __bf16* arow = A + (size_t)(row0 + mr) * lda + g * 8;
  const __bf16* brow[NT];
#pragma unroll
  for (int t = 0; t < NT; ++t)
    brow[t] = Bt + (size_t)(col0 + 16 * t + mr) * ldb + g * 8;
  for (int k0 = 0; k0 < K; k0 += 32) {
    v16bf a = frag_load_bf16(arow + k0);
    if (k0 + 32 < K) __builtin_prefetch(arow + k0 + 32, 0, 0);
#pragma unroll
    for (int t = 0; t < NT; ++t) {
      v16bf b = frag_load_bf16(brow[t] + k0);
      acc[t] = __builtin_amdgcn_wmma_f32_16x16x32_bf16(
          false, a, false, b, (short)0, acc[t], false, false);
    }
  }
}

// ---------------------------------------------------------------------------
// Kernels
// ---------------------------------------------------------------------------

// proj0/proj1 (H,D,D) -> row-major "B^T" form P*t[h][n][d] = proj[h][d][n]
__global__ void transpose_proj_kernel(const float* __restrict__ proj0,
                                      const float* __restrict__ proj1,
                                      float* __restrict__ P0t,
                                      float* __restrict__ P1t) {
  int idx = blockIdx.x * blockDim.x + threadIdx.x;
  const int per = H * D * D;
  if (idx >= 2 * per) return;
  const float* src = (idx < per) ? proj0 : proj1;
  float* dst = (idx < per) ? P0t : P1t;
  int r = idx % per;
  int h = r / (D * D);
  int rr = r % (D * D);
  int n = rr / D;   // output row of P*t = output column of proj
  int d = rr % D;
  dst[(size_t)h * D * D + (size_t)n * D + d] =
      src[(size_t)h * D * D + (size_t)d * D + n];
}

// Q/K/V = hs @ W^T, head-split store into (H,S,D)
__global__ void qkv_kernel(const float* __restrict__ hs,
                           const float* __restrict__ Wq,
                           const float* __restrict__ Wk,
                           const float* __restrict__ Wv,
                           float* __restrict__ Qr, float* __restrict__ Kr,
                           float* __restrict__ V) {
  const int wid  = (blockIdx.x * blockDim.x + threadIdx.x) >> 5;
  const int lane = threadIdx.x & 31;
  const int strips = (S / 16) * (E / 64);  // 4096 per matrix
  if (wid >= 3 * strips) return;
  const int mat = wid / strips, rem = wid % strips;
  const int row0 = (rem >> 6) * 16;
  const int col0 = (rem & 63) * 64;
  const float* W = (mat == 0) ? Wq : (mat == 1) ? Wk : Wv;
  float* dst = (mat == 0) ? Qr : (mat == 1) ? Kr : V;

  v8f acc[4] = {};
  gemm_strip_f32<4>(hs, E, row0, W, E, col0, E, lane, acc);

  const int n = lane & 15, g = lane >> 4;
#pragma unroll
  for (int t = 0; t < 4; ++t) {
    const int e0 = col0 + 16 * t;
    const int h = e0 >> 7, d0 = e0 & 127;  // 16-wide tile never straddles a head
#pragma unroll
    for (int r = 0; r < 8; ++r)
      dst[((size_t)h * S + (row0 + 8 * g + r)) * D + d0 + n] = acc[t][r];
  }
}

// In-place RoPE on (H,S,D); thread owns the (d, d+64) pair.
__global__ void rope_kernel(float* __restrict__ Qr, float* __restrict__ Kr) {
  int idx = blockIdx.x * blockDim.x + threadIdx.x;
  const int per = H * S * 64;
  if (idx >= 2 * per) return;
  float* buf = (idx < per) ? Qr : Kr;
  int r = idx % per;
  int h = r / (S * 64);
  int r2 = r % (S * 64);
  int s = r2 >> 6;
  int d = r2 & 63;
  float invf = __powf(10000.0f, -(float)d * (1.0f / 64.0f));
  float sn, c;
  __sincosf((float)s * invf, &sn, &c);
  float* p = buf + ((size_t)h * S + s) * D;
  float x1 = p[d], x2 = p[d + 64];
  p[d]      = x1 * c - x2 * sn;
  p[d + 64] = x2 * c + x1 * sn;
}

// Vt[h][d][s] = (bf16) V[h][s][d]
__global__ void vt_kernel(const float* __restrict__ V, __bf16* __restrict__ Vt) {
  int idx = blockIdx.x * blockDim.x + threadIdx.x;
  if (idx >= H * S * D) return;
  int h = idx / (S * D);
  int r = idx % (S * D);
  int s = r / D, d = r % D;
  Vt[((size_t)h * D + d) * S + s] = (__bf16)V[idx];
}

// out[h,q,k] = sum_d A[h,q,d] * Bm[h,k,d]   (used for true_score and draft_score)
__global__ void scores_kernel(const float* __restrict__ Am,
                              const float* __restrict__ Bm,
                              float* __restrict__ out) {
  const int wid  = (blockIdx.x * blockDim.x + threadIdx.x) >> 5;
  const int lane = threadIdx.x & 31;
  const int per_head = (S / 16) * (S / 64);  // 1024
  if (wid >= H * per_head) return;
  const int h = wid / per_head, rem = wid % per_head;
  const int row0 = (rem >> 4) * 16;
  const int col0 = (rem & 15) * 64;
  const float* A  = Am + (size_t)h * S * D;
  const float* Bt = Bm + (size_t)h * S * D;
  v8f acc[4] = {};
  gemm_strip_f32<4>(A, D, row0, Bt, D, col0, D, lane, acc);
  float* o = out + (size_t)h * S * S;
#pragma unroll
  for (int t = 0; t < 4; ++t) store_tile(o, S, row0, col0 + 16 * t, lane, acc[t]);
}

// tmp = silu(x @ P0t^T + b0) + x
__global__ void hashA_kernel(const float* __restrict__ x,
                             const float* __restrict__ P0t,
                             const float* __restrict__ bias0,
                             float* __restrict__ tmp) {
  const int wid  = (blockIdx.x * blockDim.x + threadIdx.x) >> 5;
  const int lane = threadIdx.x & 31;
  const int per_head = (S / 16) * (D / 64);  // 128
  if (wid >= H * per_head) return;
  const int h = wid / per_head, rem = wid % per_head;
  const int row0 = (rem >> 1) * 16;
  const int col0 = (rem & 1) * 64;
  const float* A  = x + (size_t)h * S * D;
  const float* Bt = P0t + (size_t)h * D * D;
  v8f acc[4] = {};
  gemm_strip_f32<4>(A, D, row0, Bt, D, col0, D, lane, acc);
  const int n = lane & 15, g = lane >> 4;
#pragma unroll
  for (int t = 0; t < 4; ++t) {
#pragma unroll
    for (int r = 0; r < 8; ++r) {
      int row = row0 + 8 * g + r, col = col0 + 16 * t + n;
      float v = acc[t][r] + bias0[h * D + col];
      float sl = v / (1.0f + __expf(-v));  // silu
      tmp[((size_t)h * S + row) * D + col] = sl + A[(size_t)row * D + col];
    }
  }
}

// qh = softsign(GAMMA * (tmp @ P1t^T + b1 + tmp))
__global__ void hashB_kernel(const float* __restrict__ tmp,
                             const float* __restrict__ P1t,
                             const float* __restrict__ bias1,
                             float* __restrict__ out) {
  const int wid  = (blockIdx.x * blockDim.x + threadIdx.x) >> 5;
  const int lane = threadIdx.x & 31;
  const int per_head = (S / 16) * (D / 64);
  if (wid >= H * per_head) return;
  const int h = wid / per_head, rem = wid % per_head;
  const int row0 = (rem >> 1) * 16;
  const int col0 = (rem & 1) * 64;
  const float* A  = tmp + (size_t)h * S * D;
  const float* Bt = P1t + (size_t)h * D * D;
  v8f acc[4] = {};
  gemm_strip_f32<4>(A, D, row0, Bt, D, col0, D, lane, acc);
  const int n = lane & 15, g = lane >> 4;
#pragma unroll
  for (int t = 0; t < 4; ++t) {
#pragma unroll
    for (int r = 0; r < 8; ++r) {
      int row = row0 + 8 * g + r, col = col0 + 16 * t + n;
      float v = acc[t][r] + bias1[h * D + col] + A[(size_t)row * D + col];
      float gv = v * GAMMA;
      out[((size_t)h * S + row) * D + col] = gv / (1.0f + fabsf(gv));
    }
  }
}

// causal softmax of (true_score * SCALE); writes bf16 probs (zeros past diag)
__global__ void softmax_kernel(const float* __restrict__ ts,
                               __bf16* __restrict__ probs) {
  const int bid = blockIdx.x;          // h*S + q
  const int q = bid & (S - 1);
  const int tid = threadIdx.x;
  const float* row = ts + (size_t)bid * S;
  __bf16* prow = probs + (size_t)bid * S;
  __shared__ float red[128];

  float lm = -INFINITY;
  for (int k = tid; k < S; k += 128)
    if (k <= q) lm = fmaxf(lm, row[k] * SCALE);
  red[tid] = lm;
  __syncthreads();
  for (int s = 64; s > 0; s >>= 1) {
    if (tid < s) red[tid] = fmaxf(red[tid], red[tid + s]);
    __syncthreads();
  }
  const float m = red[0];
  __syncthreads();

  float ls = 0.0f;
  for (int k = tid; k < S; k += 128)
    if (k <= q) ls += __expf(row[k] * SCALE - m);
  red[tid] = ls;
  __syncthreads();
  for (int s = 64; s > 0; s >>= 1) {
    if (tid < s) red[tid] += red[tid + s];
    __syncthreads();
  }
  const float inv = 1.0f / red[0];

  for (int k = tid; k < S; k += 128) {
    float p = (k <= q) ? __expf(row[k] * SCALE - m) * inv : 0.0f;
    prow[k] = (__bf16)p;
  }
}

// ctx[s, h*D+d] = sum_k probs[h,s,k] * Vt[h,d,k]
__global__ void pv_kernel(const __bf16* __restrict__ probs,
                          const __bf16* __restrict__ Vt,
                          float* __restrict__ ctx) {
  const int wid  = (blockIdx.x * blockDim.x + threadIdx.x) >> 5;
  const int lane = threadIdx.x & 31;
  const int per_head = (S / 16) * (D / 64);  // 128
  if (wid >= H * per_head) return;
  const int h = wid / per_head, rem = wid % per_head;
  const int row0 = (rem >> 1) * 16;
  const int col0 = (rem & 1) * 64;
  const __bf16* A  = probs + (size_t)h * S * S;
  const __bf16* Bt = Vt + (size_t)h * D * S;
  v8f acc[4] = {};
  gemm_strip_bf16<4>(A, S, row0, Bt, S, col0, S, lane, acc);
  const int n = lane & 15, g = lane >> 4;
#pragma unroll
  for (int t = 0; t < 4; ++t) {
#pragma unroll
    for (int r = 0; r < 8; ++r)
      ctx[(size_t)(row0 + 8 * g + r) * E + h * D + col0 + 16 * t + n] = acc[t][r];
  }
}

// attn = ctx @ Wo^T
__global__ void out_kernel(const float* __restrict__ ctx,
                           const float* __restrict__ Wo,
                           float* __restrict__ attn) {
  const int wid  = (blockIdx.x * blockDim.x + threadIdx.x) >> 5;
  const int lane = threadIdx.x & 31;
  const int strips = (S / 16) * (E / 64);  // 4096
  if (wid >= strips) return;
  const int row0 = (wid >> 6) * 16;
  const int col0 = (wid & 63) * 64;
  v8f acc[4] = {};
  gemm_strip_f32<4>(ctx, E, row0, Wo, E, col0, E, lane, acc);
#pragma unroll
  for (int t = 0; t < 4; ++t) store_tile(attn, E, row0, col0 + 16 * t, lane, acc[t]);
}

}  // namespace

extern "C" void kernel_launch(void* const* d_in, const int* in_sizes, int n_in,
                              void* d_out, int out_size, void* d_ws, size_t ws_size,
                              hipStream_t stream) {
  const float* hs    = (const float*)d_in[0];
  const float* Wq    = (const float*)d_in[1];
  const float* Wk    = (const float*)d_in[2];
  const float* Wv    = (const float*)d_in[3];
  const float* Wo    = (const float*)d_in[4];
  const float* proj0 = (const float*)d_in[5];
  const float* bias0 = (const float*)d_in[6];
  const float* proj1 = (const float*)d_in[7];
  const float* bias1 = (const float*)d_in[8];

  float* attn  = (float*)d_out;                    // (S, E)
  float* draft = attn + (size_t)S * E;             // (H, S, S)
  float* truth = draft + (size_t)H * S * S;        // (H, S, S)

  // Workspace layout (f32 units unless noted), ~172 MB total.
  const size_t HSD = (size_t)H * S * D;            // 4,194,304
  const size_t HDD = (size_t)H * D * D;            // 524,288
  float* Qr  = (float*)d_ws;
  float* Kr  = Qr + HSD;
  float* V   = Kr + HSD;
  float* tmp = V + HSD;                            // hash temp, later reused as ctx
  float* qh  = tmp + HSD;
  float* kh  = qh + HSD;
  float* P0t = kh + HSD;
  float* P1t = P0t + HDD;
  __bf16* Vt    = (__bf16*)(P1t + HDD);            // (H, D, S) bf16
  __bf16* probs = Vt + HSD;                        // (H, S, S) bf16

  // 1) transpose the small per-head projection matrices
  transpose_proj_kernel<<<(2 * H * D * D) / 256, 256, 0, stream>>>(proj0, proj1, P0t, P1t);
  // 2) QKV projections (3 * 4096 strips of 16x64, 8 waves/block)
  qkv_kernel<<<1536, 256, 0, stream>>>(hs, Wq, Wk, Wv, Qr, Kr, V);
  // 3) RoPE in place
  rope_kernel<<<(2 * H * S * 64) / 256, 256, 0, stream>>>(Qr, Kr);
  // 4) V -> V^T (bf16)
  vt_kernel<<<(H * S * D) / 256, 256, 0, stream>>>(V, Vt);
  // 5) true_score = Qr @ Kr^T  (straight to output)
  scores_kernel<<<4096, 256, 0, stream>>>(Qr, Kr, truth);
  // 6) hash(Qr) -> qh ; hash(Kr) -> kh
  hashA_kernel<<<512, 256, 0, stream>>>(Qr, P0t, bias0, tmp);
  hashB_kernel<<<512, 256, 0, stream>>>(tmp, P1t, bias1, qh);
  hashA_kernel<<<512, 256, 0, stream>>>(Kr, P0t, bias0, tmp);
  hashB_kernel<<<512, 256, 0, stream>>>(tmp, P1t, bias1, kh);
  // 7) draft_score = qh @ kh^T (straight to output)
  scores_kernel<<<4096, 256, 0, stream>>>(qh, kh, draft);
  // 8) causal softmax of SCALE*true_score -> bf16 probs
  softmax_kernel<<<H * S, 128, 0, stream>>>(truth, probs);
  // 9) ctx = probs @ V  (tmp reused as ctx, (S,E) layout)
  pv_kernel<<<512, 256, 0, stream>>>(probs, Vt, tmp);
  // 10) attn = ctx @ Wo^T
  out_kernel<<<512, 256, 0, stream>>>(tmp, Wo, attn);
}
